// WindowAttention_86251533238310
// MI455X (gfx1250) — matrix operand order repaired and checked by reading, not compiled, not run on hardware
//
#include <hip/hip_runtime.h>

typedef __attribute__((ext_vector_type(16))) __bf16 v16bf;
typedef __attribute__((ext_vector_type(8)))  float  v8f;
typedef __attribute__((ext_vector_type(4)))  int    v4i;

union FragBF { v16bf v; unsigned u[8]; };

__device__ __forceinline__ unsigned short f2bf(float x) {
  unsigned u = __builtin_bit_cast(unsigned, x);
  unsigned r = (u + 0x7FFFu + ((u >> 16) & 1u)) >> 16;   // round-to-nearest-even
  return (unsigned short)r;
}

// Single-instruction lane-xor shuffle via ds_swizzle (group-of-32 mode):
// lane_out = ((lane & 0x1f) | 0) ^ m  -> stays inside each 16-lane half for m < 16.
#define SWZ_XOR(x, m)                                                          \
  __builtin_bit_cast(float, __builtin_amdgcn_ds_swizzle(                       \
      __builtin_bit_cast(int, (x)), ((m) << 10) | 0x1f))

__device__ __forceinline__ float red16_add(float v) {
  v += SWZ_XOR(v, 1); v += SWZ_XOR(v, 2); v += SWZ_XOR(v, 4); v += SWZ_XOR(v, 8);
  return v;
}
__device__ __forceinline__ float red16_max(float v) {
  v = fmaxf(v, SWZ_XOR(v, 1)); v = fmaxf(v, SWZ_XOR(v, 2));
  v = fmaxf(v, SWZ_XOR(v, 4)); v = fmaxf(v, SWZ_XOR(v, 8));
  return v;
}

static constexpr int Bn   = 2048;
static constexpr int Nw   = 49;
static constexpr int C    = 512;
static constexpr int H    = 16;
static constexpr int Mtot = Bn * Nw;           // 100352 = 784 * 128
static constexpr float LOGIT_MAX = 4.6051702f; // ln(100)

// ---- CDNA5 async global->LDS copy (guarded; manual fallback keeps compile safe) ----
#if __has_builtin(__builtin_amdgcn_global_load_async_to_lds_b128)
#define HAVE_ASYNC_LDS 1
typedef __attribute__((address_space(1))) v4i gas_v4i;
typedef __attribute__((address_space(3))) v4i las_v4i;
__device__ __forceinline__ void async_copy_b128(const void* g, void* l) {
  __builtin_amdgcn_global_load_async_to_lds_b128((gas_v4i*)g, (las_v4i*)l, 0, 0);
}
#else
#define HAVE_ASYNC_LDS 0
#endif

__device__ __forceinline__ void wait_async0() {
#if __has_builtin(__builtin_amdgcn_s_wait_asynccnt)
  __builtin_amdgcn_s_wait_asynccnt(0);
#else
  asm volatile("s_wait_asynccnt 0x0" ::: "memory");
#endif
}

// ---------------------------------------------------------------------------
// Prep kernels
// ---------------------------------------------------------------------------
__global__ void cvt_weights(const float* __restrict__ wqkv, const float* __restrict__ projw,
                            unsigned short* __restrict__ wqkv_bf, unsigned short* __restrict__ projw_bf) {
  int id = blockIdx.x * blockDim.x + threadIdx.x;
  const int n0 = C * 3 * C, n1 = C * C;
  if (id < n0)           wqkv_bf[id] = f2bf(wqkv[id]);
  else if (id < n0 + n1) projw_bf[id - n0] = f2bf(projw[id - n0]);
}

__device__ __forceinline__ float cpb_coord(int v) {
  float x = (float)v * (8.0f / 6.0f);
  float l = __log2f(fabsf(x) + 1.0f) * (1.0f / 3.0f);
  return copysignf(l, x);
}

__global__ void cpb_table(const float* __restrict__ w1, const float* __restrict__ b1,
                          const float* __restrict__ w2, float* __restrict__ table_out) {
  int ri = blockIdx.x, l = threadIdx.x;
  float t0 = cpb_coord(ri / 13 - 6);
  float t1 = cpb_coord(ri % 13 - 6);
  float ph[16];
#pragma unroll
  for (int h = 0; h < 16; ++h) ph[h] = 0.f;
  for (int j = l; j < 512; j += 32) {
    float hid = fmaxf(t0 * w1[j] + t1 * w1[512 + j] + b1[j], 0.f);
#pragma unroll
    for (int h = 0; h < 16; ++h) ph[h] += hid * w2[j * 16 + h];
  }
#pragma unroll
  for (int h = 0; h < 16; ++h) {
    float v = ph[h];
    v += SWZ_XOR(v, 16);
    ph[h] = red16_add(v);
  }
  if (l < 16) table_out[ri * 16 + l] = ph[l];
}

// Padded bias table [H][64][64] with key-mask baked in + per-head logit scale.
__global__ void bias_expand(const float* __restrict__ table, const float* __restrict__ logit_scale,
                            float* __restrict__ biasPad, float* __restrict__ scalef) {
  int id = blockIdx.x * blockDim.x + threadIdx.x;
  if (id < 16) scalef[id] = __expf(fminf(logit_scale[id], LOGIT_MAX));
  if (id < H * 64 * 64) {
    int h = id >> 12, qk = id & 4095, q = qk >> 6, key = qk & 63;
    float v;
    if (key >= Nw)     v = -1e30f;                 // masked padded keys
    else if (q >= Nw)  v = 0.f;                    // padded queries (discarded)
    else {
      int idx = (q / 7 - key / 7 + 6) * 13 + (q % 7 - key % 7 + 6);
      v = 16.f / (1.f + __expf(-table[idx * 16 + h]));
    }
    biasPad[id] = v;
  }
}

// ---------------------------------------------------------------------------
// WMMA GEMM: 128x64 block tile, 8 waves, 32x32 tile/wave, double-buffered LDS,
// software pipelined (stage next K-tile before computing current).
// MODE 0: A = x fp32 (cvt on stage); fused q/v bias + L2 norm epilogue -> bf16
// MODE 1: A = attn-out bf16 (async staged); +proj_b -> fp32
// ---------------------------------------------------------------------------
template<int MODE, int NOUT>
__global__ __launch_bounds__(256) void gemm_wmma(
    const float* __restrict__ Afp, const unsigned short* __restrict__ Abf,
    const unsigned short* __restrict__ Bw,
    const float* __restrict__ bias0, const float* __restrict__ bias1,
    unsigned short* __restrict__ outbf, float* __restrict__ outf) {
  __shared__ __align__(16) unsigned short As[2][128 * 32];  // [row][k]
  __shared__ __align__(16) unsigned short Bs[2][64 * 32];   // [col][k]

  const int t = threadIdx.x, lane = t & 31, wave = t >> 5;
  const int m0 = blockIdx.x * 128, n0 = blockIdx.y * 64;
  const int rb = (wave & 3) * 32;
  const int cb = (wave >> 2) * 32;
  const int l16   = lane & 15;
  const int koffA = (lane < 16) ? 0 : 8;
  const int koffB = (lane < 16) ? 0 : 8;

  v8f c00 = {}, c01 = {}, c10 = {}, c11 = {};

  auto stageA = [&](int kk, int buf) {
    if (MODE == 0) {                       // fp32 -> bf16, float4 vectorized
#pragma unroll
      for (int i = 0; i < 4; ++i) {
        int cidx = i * 256 + t;            // 1024 float4 chunks
        int row = cidx >> 3, k4 = cidx & 7;
        float4 f = reinterpret_cast<const float4*>(Afp + (size_t)(m0 + row) * C + kk)[k4];
        unsigned lo = (unsigned)f2bf(f.x) | ((unsigned)f2bf(f.y) << 16);
        unsigned hi = (unsigned)f2bf(f.z) | ((unsigned)f2bf(f.w) << 16);
        ((uint2*)As[buf])[row * 8 + k4] = make_uint2(lo, hi);
      }
    } else {
#if HAVE_ASYNC_LDS
#pragma unroll
      for (int i = 0; i < 2; ++i) {
        int cidx = i * 256 + t;            // 512 b128 chunks
        int row = cidx >> 2, p = cidx & 3;
        async_copy_b128((const char*)(Abf + (size_t)(m0 + row) * C + kk) + p * 16,
                        (char*)As[buf] + cidx * 16);
      }
#else
#pragma unroll
      for (int i = 0; i < 2; ++i) {
        int cidx = i * 256 + t;
        int row = cidx >> 2, p = cidx & 3;
        *(uint4*)((char*)As[buf] + cidx * 16) =
            *(const uint4*)((const char*)(Abf + (size_t)(m0 + row) * C + kk) + p * 16);
      }
#endif
    }
  };
  auto stageB = [&](int kk, int buf) {     // B: [k][col] global -> [col][k] LDS
#pragma unroll
    for (int i = 0; i < 8; ++i) {
      int e = i * 256 + t;
      int k = e >> 6, col = e & 63;
      Bs[buf][col * 32 + k] = Bw[(size_t)(kk + k) * NOUT + n0 + col];
    }
  };

  stageA(0, 0);
  stageB(0, 0);
#if HAVE_ASYNC_LDS
  if (MODE == 1) wait_async0();
#endif
  __syncthreads();

  for (int s = 0; s < C / 32; ++s) {
    const int cur = s & 1;
    if (s + 1 < C / 32) {                  // start next tile's staging early
      stageA((s + 1) * 32, 1 - cur);
      stageB((s + 1) * 32, 1 - cur);
    }

    const unsigned* As32 = (const unsigned*)As[cur];
    const unsigned* Bs32 = (const unsigned*)Bs[cur];
    FragBF a0, a1, b0, b1;
#pragma unroll
    for (int r = 0; r < 8; ++r) {
      int rr = r;
      a0.u[r] = As32[(rb +      l16) * 16 + koffA + rr];
      a1.u[r] = As32[(rb + 16 + l16) * 16 + koffA + rr];
      b0.u[r] = Bs32[(cb +      l16) * 16 + koffB + r];
      b1.u[r] = Bs32[(cb + 16 + l16) * 16 + koffB + r];
    }
    c00 = __builtin_amdgcn_wmma_f32_16x16x32_bf16(false, a0.v, false, b0.v, (short)0, c00, false, false);
    c01 = __builtin_amdgcn_wmma_f32_16x16x32_bf16(false, a0.v, false, b1.v, (short)0, c01, false, false);
    c10 = __builtin_amdgcn_wmma_f32_16x16x32_bf16(false, a1.v, false, b0.v, (short)0, c10, false, false);
    c11 = __builtin_amdgcn_wmma_f32_16x16x32_bf16(false, a1.v, false, b1.v, (short)0, c11, false, false);

#if HAVE_ASYNC_LDS
    if (MODE == 1) wait_async0();          // own async writes done before barrier
#endif
    __syncthreads();
  }

  const int gcb  = n0 + cb;
  const int col0 = gcb + l16, col1 = gcb + 16 + l16;
  v8f* ca[2][2] = {{&c00, &c01}, {&c10, &c11}};

#pragma unroll
  for (int hf = 0; hf < 2; ++hf) {
    v8f& x0 = *ca[hf][0];
    v8f& x1 = *ca[hf][1];
    if (MODE == 0) {
      if (gcb < C) {                       // q: +q_bias, then L2 norm
        float qb0 = bias0[col0], qb1 = bias0[col1];
#pragma unroll
        for (int r = 0; r < 8; ++r) { x0[r] += qb0; x1[r] += qb1; }
      } else if (gcb >= 2 * C) {           // v: +v_bias
        float vb0 = bias1[col0 - 2 * C], vb1 = bias1[col1 - 2 * C];
#pragma unroll
        for (int r = 0; r < 8; ++r) { x0[r] += vb0; x1[r] += vb1; }
      }
      if (gcb < 2 * C) {                   // q or k: row L2 norm over the head's 32 cols
#pragma unroll
        for (int r = 0; r < 8; ++r) {
          float ssum = red16_add(x0[r] * x0[r] + x1[r] * x1[r]);
          float inv = rsqrtf(fmaxf(ssum, 1e-12f));
          x0[r] *= inv; x1[r] *= inv;
        }
      }
#pragma unroll
      for (int r = 0; r < 8; ++r) {
        int row = m0 + rb + hf * 16 + ((lane < 16) ? r : r + 8);
        outbf[(size_t)row * NOUT + col0] = f2bf(x0[r]);
        outbf[(size_t)row * NOUT + col1] = f2bf(x1[r]);
      }
    } else {
      float pb0 = bias0[col0], pb1 = bias0[col1];
#pragma unroll
      for (int r = 0; r < 8; ++r) {
        int row = m0 + rb + hf * 16 + ((lane < 16) ? r : r + 8);
        outf[(size_t)row * NOUT + col0] = x0[r] + pb0;
        outf[(size_t)row * NOUT + col1] = x1[r] + pb1;
      }
    }
  }
}

// ---------------------------------------------------------------------------
// Fused attention per (window, head). 4 waves; each owns 16 query rows.
// ---------------------------------------------------------------------------
__global__ __launch_bounds__(128) void attn_wmma(
    const unsigned short* __restrict__ qkv, const float* __restrict__ biasPad,
    const float* __restrict__ scalef, unsigned short* __restrict__ outbf) {
  __shared__ __align__(16) unsigned short Qs[64 * 32];  // [q][d]
  __shared__ __align__(16) unsigned short Ks[64 * 32];  // [key][d]
  __shared__ __align__(16) unsigned short Vs[32 * 64];  // [d][key]
  __shared__ __align__(16) unsigned short Ps[64 * 64];  // [q][key]
  const unsigned* Qs32 = (const unsigned*)Qs;
  const unsigned* Ks32 = (const unsigned*)Ks;
  const unsigned* Vs32 = (const unsigned*)Vs;
  const unsigned* Ps32 = (const unsigned*)Ps;

  const int bh = blockIdx.x, b = bh >> 4, h = bh & 15;
  const int t = threadIdx.x, lane = t & 31, wave = t >> 5;
  const int l16   = lane & 15;
  const int koffA = (lane < 16) ? 0 : 8;
  const int koffB = (lane < 16) ? 0 : 8;

  const unsigned short* baseQ = qkv + (size_t)b * Nw * (3 * C) + h * 32;

  // zero padded rows/keys (stale LDS could hold Inf-pattern bf16 -> NaN via 0*Inf)
  for (int e = t; e < 64 * 32; e += 128) {
    if ((e >> 5) >= Nw) { Qs[e] = 0; Ks[e] = 0; }
  }
  for (int e = t; e < 32 * 64; e += 128)
    if ((e & 63) >= Nw) Vs[e] = 0;

#if HAVE_ASYNC_LDS
  for (int cidx = t; cidx < Nw * 4; cidx += 128) {       // 49 rows * 4 x b128
    int row = cidx >> 2, p = cidx & 3;
    const char* g = (const char*)(baseQ + (size_t)row * (3 * C)) + p * 16;
    async_copy_b128(g,         (char*)Qs + cidx * 16);
    async_copy_b128(g + 2 * C, (char*)Ks + cidx * 16);   // +C ushorts
  }
#else
  for (int e = t; e < Nw * 32; e += 128) {
    int row = e >> 5, dd = e & 31;
    const unsigned short* rp = baseQ + (size_t)row * (3 * C) + dd;
    Qs[row * 32 + dd] = rp[0];
    Ks[row * 32 + dd] = rp[C];
  }
#endif
  for (int e = t; e < Nw * 32; e += 128) {               // V with transpose
    int row = e >> 5, dd = e & 31;
    Vs[dd * 64 + row] = baseQ[(size_t)row * (3 * C) + 2 * C + dd];
  }
#if HAVE_ASYNC_LDS
  wait_async0();
#endif
  __syncthreads();

  // ---- S = Qn @ Kn^T ----
  FragBF aq;
#pragma unroll
  for (int r = 0; r < 8; ++r) {
    int rr = r;
    aq.u[r] = Qs32[(wave * 16 + l16) * 16 + koffA + rr];
  }
  v8f acc[4];
#pragma unroll
  for (int kt = 0; kt < 4; ++kt) {
    FragBF bk;
#pragma unroll
    for (int r = 0; r < 8; ++r)
      bk.u[r] = Ks32[(kt * 16 + l16) * 16 + koffB + r];
    v8f z = {};
    acc[kt] = __builtin_amdgcn_wmma_f32_16x16x32_bf16(false, aq.v, false, bk.v, (short)0, z, false, false);
  }

  // ---- scale + padded bias (mask baked in), branchless ----
  const float sc = scalef[h];
  const float* bp = biasPad + (size_t)h * 4096;
#pragma unroll
  for (int kt = 0; kt < 4; ++kt) {
#pragma unroll
    for (int r = 0; r < 8; ++r) {
      int q   = wave * 16 + ((lane < 16) ? r : r + 8);
      int key = kt * 16 + l16;
      acc[kt][r] = acc[kt][r] * sc + bp[q * 64 + key];
    }
  }

  // ---- softmax over 64 keys (single ds_swizzle per reduce step) ----
#pragma unroll
  for (int r = 0; r < 8; ++r) {
    float m = red16_max(fmaxf(fmaxf(acc[0][r], acc[1][r]), fmaxf(acc[2][r], acc[3][r])));
    float s = 0.f;
#pragma unroll
    for (int kt = 0; kt < 4; ++kt) { float e = __expf(acc[kt][r] - m); acc[kt][r] = e; s += e; }
    s = red16_add(s);
    float inv = 1.f / s;
#pragma unroll
    for (int kt = 0; kt < 4; ++kt) acc[kt][r] *= inv;
  }

  // ---- P: C layout -> LDS -> A layout ----
#pragma unroll
  for (int kt = 0; kt < 4; ++kt)
#pragma unroll
    for (int r = 0; r < 8; ++r) {
      int q   = wave * 16 + ((lane < 16) ? r : r + 8);
      int key = kt * 16 + l16;
      Ps[q * 64 + key] = f2bf(acc[kt][r]);
    }
  __syncthreads();

  FragBF p0, p1;
#pragma unroll
  for (int r = 0; r < 8; ++r) {
    int rr = r;
    int qrow = wave * 16 + l16;
    p0.u[r] = Ps32[qrow * 32 +  0 + koffA + rr];
    p1.u[r] = Ps32[qrow * 32 + 16 + koffA + rr];
  }

  // ---- O = P @ V ----
#pragma unroll
  for (int dt = 0; dt < 2; ++dt) {
    v8f o = {};
#pragma unroll
    for (int kc = 0; kc < 2; ++kc) {
      FragBF bv;
#pragma unroll
      for (int r = 0; r < 8; ++r)
        bv.u[r] = Vs32[(dt * 16 + l16) * 32 + kc * 16 + koffB + r];
      o = __builtin_amdgcn_wmma_f32_16x16x32_bf16(false, (kc == 0) ? p0.v : p1.v,
                                                  false, bv.v, (short)0, o, false, false);
    }
#pragma unroll
    for (int r = 0; r < 8; ++r) {
      int q = wave * 16 + ((lane < 16) ? r : r + 8);
      if (q < Nw)
        outbf[((size_t)b * Nw + q) * C + h * 32 + dt * 16 + l16] = f2bf(o[r]);
    }
  }
}

// ---------------------------------------------------------------------------
extern "C" void kernel_launch(void* const* d_in, const int* in_sizes, int n_in,
                              void* d_out, int out_size, void* d_ws, size_t ws_size,
                              hipStream_t stream) {
  (void)in_sizes; (void)n_in; (void)out_size; (void)ws_size;
  const float* x        = (const float*)d_in[0];
  const float* w_qkv    = (const float*)d_in[1];
  const float* q_bias   = (const float*)d_in[2];
  const float* v_bias   = (const float*)d_in[3];
  const float* logit_sc = (const float*)d_in[4];
  const float* cpb_w1   = (const float*)d_in[5];
  const float* cpb_b1   = (const float*)d_in[6];
  const float* cpb_w2   = (const float*)d_in[7];
  const float* proj_w   = (const float*)d_in[8];
  const float* proj_b   = (const float*)d_in[9];
  float* out = (float*)d_out;

  char* ws = (char*)d_ws;
  size_t off = 0;
  auto take = [&](size_t bytes) -> void* {
    void* p = ws + off;
    off = (off + bytes + 255) & ~(size_t)255;
    return p;
  };
  unsigned short* wqkv_bf  = (unsigned short*)take((size_t)C * 3 * C * 2);
  unsigned short* projw_bf = (unsigned short*)take((size_t)C * C * 2);
  float* tablef  = (float*)take(169 * 16 * 4);
  float* biasPad = (float*)take((size_t)H * 64 * 64 * 4);
  float* scalef  = (float*)take(64);
  unsigned short* qkv_bf = (unsigned short*)take((size_t)Mtot * 3 * C * 2);
  unsigned short* ao_bf  = (unsigned short*)take((size_t)Mtot * C * 2);

  cvt_weights<<<4096, 256, 0, stream>>>(w_qkv, proj_w, wqkv_bf, projw_bf);
  cpb_table<<<169, 32, 0, stream>>>(cpb_w1, cpb_b1, cpb_w2, tablef);
  bias_expand<<<256, 256, 0, stream>>>(tablef, logit_sc, biasPad, scalef);

  gemm_wmma<0, 3 * C><<<dim3(Mtot / 128, (3 * C) / 64), 256, 0, stream>>>(
      x, nullptr, wqkv_bf, q_bias, v_bias, qkv_bf, nullptr);

  attn_wmma<<<Bn * H, 128, 0, stream>>>(qkv_bf, biasPad, scalef, ao_bf);

  gemm_wmma<1, C><<<dim3(Mtot / 128, C / 64), 256, 0, stream>>>(
      nullptr, ao_bf, projw_bf, proj_b, nullptr, nullptr, out);
}
